// RegNet_58360015618540
// MI455X (gfx1250) — matrix operand order repaired and checked by reading, not compiled, not run on hardware
//
#include <hip/hip_runtime.h>
#include <hip/hip_bf16.h>

typedef __attribute__((ext_vector_type(16))) _Float16 v16h;
typedef __attribute__((ext_vector_type(8)))  _Float16 v8h;
typedef __attribute__((ext_vector_type(4)))  _Float16 v4h;
typedef __attribute__((ext_vector_type(8)))  float    v8f;

union Frag { v16h v; v8h h[2]; };

#define TPB 256

// ---------------------------------------------------------------------------
// LDS-tiled generic-stride f16 x f16 -> f32 GEMM on V_WMMA_F32_16X16X32_F16.
// S[m,n] = sum_k A(m,k)*B(k,n); strides in elements; batch via blockIdx.z.
// Block: 256 threads = 8 waves; macro tile 128(M) x 64(N); wave tile 32x32
// (4 WMMAs/chunk, fragments reused across two tiles each).
// Requirements: M % 128 == 0, N % 64 == 0, K % 32 == 0 (buffers padded).
// ---------------------------------------------------------------------------
__global__ __launch_bounds__(256) void wmma_gemm(
    const _Float16* __restrict__ A, const _Float16* __restrict__ Bm,
    float* __restrict__ Cm, int M, int Nn, int Kk,
    long sam, long sak, long abatch,
    long sbk, long sbn, long bbatch,
    long scm, long scn, long cbatch)
{
  (void)M; (void)Nn;
  // rows padded to 40 halves (80B) to spread fragment reads across banks
  __shared__ __align__(16) _Float16 As[128 * 40];
  __shared__ __align__(16) _Float16 Bs[64 * 40];

  const int lane = threadIdx.x;            // 0..31 (wave32)
  const int w    = threadIdx.y;            // 0..7
  const int tid  = w * 32 + lane;
  const int b    = blockIdx.z;
  const int mBase = blockIdx.x * 128;
  const int nBase = blockIdx.y * 64;
  const int wm = w >> 1, wn = w & 1;       // wave sub-tile origin (wm*32, wn*32)
  const int hi = lane >> 4, ml = lane & 15;

  const _Float16* __restrict__ Ag = A  + (long)b * abatch + (long)mBase * sam;
  const _Float16* __restrict__ Bg = Bm + (long)b * bbatch + (long)nBase * sbn;

  v8f acc00 = {}, acc01 = {}, acc10 = {}, acc11 = {};

  for (int kc = 0; kc < Kk; kc += 32) {
    __syncthreads();
    // ---- stage A tile (128 x 32), LDS layout As[m*40 + k] ----
    if (sak == 1) {                                    // k-contiguous in memory
      for (int e = tid; e < 1024; e += 256) {
        int m = e >> 3, k4 = (e & 7) << 2;
        *(v4h*)&As[m * 40 + k4] = *(const v4h*)(Ag + (long)m * sam + (kc + k4));
      }
    } else if (sam == 1 && (sak & 3) == 0) {           // m-contiguous in memory
      for (int e = tid; e < 1024; e += 256) {
        int m4 = (e & 31) << 2, k = e >> 5;
        v4h v = *(const v4h*)(Ag + m4 + (long)(kc + k) * sak);
        As[(m4 + 0) * 40 + k] = v[0];
        As[(m4 + 1) * 40 + k] = v[1];
        As[(m4 + 2) * 40 + k] = v[2];
        As[(m4 + 3) * 40 + k] = v[3];
      }
    } else {                                           // generic fallback
      for (int e = tid; e < 4096; e += 256) {
        int m = e >> 5, k = e & 31;
        As[m * 40 + k] = Ag[(long)m * sam + (long)(kc + k) * sak];
      }
    }
    // ---- stage B tile (32 x 64), LDS layout Bs[n*40 + k] (k-contig/col) ----
    if (sbk == 1) {
      for (int e = tid; e < 512; e += 256) {
        int n = e >> 3, k4 = (e & 7) << 2;
        *(v4h*)&Bs[n * 40 + k4] = *(const v4h*)(Bg + (long)n * sbn + (kc + k4));
      }
    } else if (sbn == 1 && (sbk & 3) == 0) {
      for (int e = tid; e < 512; e += 256) {
        int n4 = (e & 15) << 2, k = e >> 4;
        v4h v = *(const v4h*)(Bg + n4 + (long)(kc + k) * sbk);
        Bs[(n4 + 0) * 40 + k] = v[0];
        Bs[(n4 + 1) * 40 + k] = v[1];
        Bs[(n4 + 2) * 40 + k] = v[2];
        Bs[(n4 + 3) * 40 + k] = v[3];
      }
    } else {
      for (int e = tid; e < 2048; e += 256) {
        int n = e >> 5, k = e & 31;
        Bs[n * 40 + k] = Bg[(long)(kc + k) * sbk + (long)n * sbn];
      }
    }
    __syncthreads();
    // ---- fragments via 16B LDS loads, per documented CDNA5 layouts ----
    // A (16x32): lanes 0-15 hold K {hi*8..+7, 16+hi*8..+7} for row M=lane&15
    // B (32x16): lanes hold K {hi*8..+7, 16+hi*8..+7} for col N=lane&15
    Frag a0, a1, b0, b1;
    const int ar0 = (wm * 32 + ml) * 40;
    const int ar1 = (wm * 32 + 16 + ml) * 40;
    a0.h[0] = *(const v8h*)&As[ar0 + hi * 8];
    a0.h[1] = *(const v8h*)&As[ar0 + 16 + hi * 8];
    a1.h[0] = *(const v8h*)&As[ar1 + hi * 8];
    a1.h[1] = *(const v8h*)&As[ar1 + 16 + hi * 8];
    const int bc0 = (wn * 32 + ml) * 40;
    const int bc1 = (wn * 32 + 16 + ml) * 40;
    b0.h[0] = *(const v8h*)&Bs[bc0 + hi * 8];
    b0.h[1] = *(const v8h*)&Bs[bc0 + 16 + hi * 8];
    b1.h[0] = *(const v8h*)&Bs[bc1 + hi * 8];
    b1.h[1] = *(const v8h*)&Bs[bc1 + 16 + hi * 8];
    acc00 = __builtin_amdgcn_wmma_f32_16x16x32_f16(false, a0.v, false, b0.v, (short)0, acc00, false, false);
    acc01 = __builtin_amdgcn_wmma_f32_16x16x32_f16(false, a0.v, false, b1.v, (short)0, acc01, false, false);
    acc10 = __builtin_amdgcn_wmma_f32_16x16x32_f16(false, a1.v, false, b0.v, (short)0, acc10, false, false);
    acc11 = __builtin_amdgcn_wmma_f32_16x16x32_f16(false, a1.v, false, b1.v, (short)0, acc11, false, false);
  }
  // ---- store: D VGPR r -> row M = r + 8*hi, col N = lane&15 ----
  float* __restrict__ Cp = Cm + (long)b * cbatch;
  const int mw = mBase + wm * 32, nw = nBase + wn * 32;
#pragma unroll
  for (int r = 0; r < 8; ++r) {
    const int mr = r + hi * 8;
    Cp[(long)(mw + mr) * scm      + (long)(nw + ml) * scn]      = acc00[r];
    Cp[(long)(mw + mr) * scm      + (long)(nw + 16 + ml) * scn] = acc01[r];
    Cp[(long)(mw + 16 + mr) * scm + (long)(nw + ml) * scn]      = acc10[r];
    Cp[(long)(mw + 16 + mr) * scm + (long)(nw + 16 + ml) * scn] = acc11[r];
  }
}

// ---------------------------------------------------------------------------
// Column L2-normalize (B,C,N) f32 -> f16 normalized (and optional raw f16)
// ---------------------------------------------------------------------------
__global__ void colnorm(const float* __restrict__ x, _Float16* __restrict__ raw16,
                        _Float16* __restrict__ n16, int C, int N, int total)
{
  int i = blockIdx.x * blockDim.x + threadIdx.x;
  if (i >= total) return;
  int b = i / N, n = i - b * N;
  const float* col = x + (long)b * C * N + n;
  float ss = 0.f;
  for (int c = 0; c < C; ++c) { float v = col[(long)c * N]; ss += v * v; }
  float inv = 1.f / fmaxf(sqrtf(ss), 1e-12f);
  for (int c = 0; c < C; ++c) {
    float v = col[(long)c * N];
    long o = (long)b * C * N + (long)c * N + n;
    if (raw16) raw16[o] = (_Float16)v;
    n16[o] = (_Float16)(v * inv);
  }
}

__global__ void f32_to_f16(const float* __restrict__ in, _Float16* __restrict__ out, long n)
{
  long i = (long)blockIdx.x * blockDim.x + threadIdx.x;
  if (i < n) out[i] = (_Float16)in[i];
}

// softmax over a strided row of S (f32) -> contiguous f16 probabilities
__global__ void softmax_f16(const float* __restrict__ S, _Float16* __restrict__ P,
                            int len, long rs, long es, long sbatch, long obatch)
{
  __shared__ float red[TPB];
  const int r = blockIdx.x, b = blockIdx.y, tid = threadIdx.x;
  const float* base = S + (long)b * sbatch + (long)r * rs;
  float mx = -3.4e38f;
  for (int j = tid; j < len; j += TPB) mx = fmaxf(mx, base[(long)j * es]);
  red[tid] = mx; __syncthreads();
  for (int s = TPB / 2; s > 0; s >>= 1) { if (tid < s) red[tid] = fmaxf(red[tid], red[tid + s]); __syncthreads(); }
  mx = red[0]; __syncthreads();
  float sum = 0.f;
  for (int j = tid; j < len; j += TPB) sum += __expf(base[(long)j * es] - mx);
  red[tid] = sum; __syncthreads();
  for (int s = TPB / 2; s > 0; s >>= 1) { if (tid < s) red[tid] += red[tid + s]; __syncthreads(); }
  float inv = 1.f / red[0];
  _Float16* out = P + (long)b * obatch + (long)r * len;
  for (int j = tid; j < len; j += TPB) out[j] = (_Float16)(__expf(base[(long)j * es] - mx) * inv);
}

// max over a strided row
__global__ void rowmax(const float* __restrict__ X, float* __restrict__ out,
                       int len, long rs, long es, long sbatch, int rows)
{
  __shared__ float red[TPB];
  const int r = blockIdx.x, b = blockIdx.y, tid = threadIdx.x;
  const float* base = X + (long)b * sbatch + (long)r * rs;
  float mx = -3.4e38f;
  for (int j = tid; j < len; j += TPB) mx = fmaxf(mx, base[(long)j * es]);
  red[tid] = mx; __syncthreads();
  for (int s = TPB / 2; s > 0; s >>= 1) { if (tid < s) red[tid] = fmaxf(red[tid], red[tid + s]); __syncthreads(); }
  if (tid == 0) out[(long)b * rows + r] = red[0];
}

// feat = concat([sp, glob_tar, gmax, glob_tar - gmax]) along channels, f16
__global__ void assemble_feat(const _Float16* __restrict__ raw16, const float* __restrict__ gt,
                              const float* __restrict__ gmax, _Float16* __restrict__ feat,
                              int C1, int N, long total)
{
  long i = (long)blockIdx.x * blockDim.x + threadIdx.x;
  if (i >= total) return;
  int n = (int)(i % N);
  long t = i / N;
  int ch = (int)(t % (4 * C1));
  int b  = (int)(t / (4 * C1));
  float v;
  if (ch < C1)          v = (float)raw16[((long)b * C1 + ch) * N + n];
  else if (ch < 2 * C1) v = gt[((long)b * C1 + (ch - C1)) * N + n];
  else if (ch < 3 * C1) v = gmax[(long)b * C1 + (ch - 2 * C1)];
  else                  v = gt[((long)b * C1 + (ch - 3 * C1)) * N + n]
                            - gmax[(long)b * C1 + (ch - 3 * C1)];
  feat[i] = (_Float16)v;
}

// BatchNorm batch statistics over (B,N) per channel
__global__ void bn_stats(const float* __restrict__ y, float* __restrict__ mu,
                         float* __restrict__ var, int Bn, int C, int N)
{
  __shared__ float r1[TPB], r2[TPB];
  const int o = blockIdx.x, tid = threadIdx.x;
  const long tot = (long)Bn * N;
  float s = 0.f, s2 = 0.f;
  for (long i = tid; i < tot; i += TPB) {
    long b = i / N, n = i - b * N;
    float v = y[(b * C + o) * (long)N + n];
    s += v; s2 += v * v;
  }
  r1[tid] = s; r2[tid] = s2; __syncthreads();
  for (int st = TPB / 2; st > 0; st >>= 1) {
    if (tid < st) { r1[tid] += r1[tid + st]; r2[tid] += r2[tid + st]; }
    __syncthreads();
  }
  if (tid == 0) { float m = r1[0] / tot; mu[o] = m; var[o] = fmaxf(r2[0] / tot - m * m, 0.f); }
}

// BN-normalize + LeakyReLU (conv bias cancels under batch-stat BN)
__global__ void bn_lrelu(const float* __restrict__ y, const float* __restrict__ mu,
                         const float* __restrict__ var, const float* __restrict__ g,
                         const float* __restrict__ be, _Float16* out16, float* out32,
                         int C, int N, long total)
{
  long i = (long)blockIdx.x * blockDim.x + threadIdx.x;
  if (i >= total) return;
  int o = (int)((i / N) % C);
  float v = (y[i] - mu[o]) * rsqrtf(var[o] + 1e-5f) * g[o] + be[o];
  v = v > 0.f ? v : 0.01f * v;
  if (out16) out16[i] = (_Float16)v;
  if (out32) out32[i] = v;
}

// per-batch top-K selection via in-LDS bitonic sort (N = 2048, pow2)
__global__ __launch_bounds__(1024) void topk_sort(const float* __restrict__ score,
                                                  int* __restrict__ idx_out, int N, int K)
{
  __shared__ unsigned long long keys[2048];
  const int b = blockIdx.x, tid = threadIdx.x;
  for (int i = tid; i < N; i += 1024) {
    unsigned u = __float_as_uint(score[(long)b * N + i]);
    u = (u & 0x80000000u) ? ~u : (u | 0x80000000u);          // ascending float map
    keys[i] = ((unsigned long long)(~u) << 32) | (unsigned)i; // desc score, tie->low idx
  }
  __syncthreads();
  for (int ksz = 2; ksz <= N; ksz <<= 1)
    for (int j = ksz >> 1; j > 0; j >>= 1) {
      for (int i = tid; i < N; i += 1024) {
        int l = i ^ j;
        if (l > i) {
          bool up = ((i & ksz) == 0);
          unsigned long long a = keys[i], c = keys[l];
          if ((a > c) == up) { keys[i] = c; keys[l] = a; }
        }
      }
      __syncthreads();
    }
  for (int i = tid; i < K; i += 1024) idx_out[(long)b * K + i] = (int)(keys[i] & 0xffffffffu);
}

// gather embedding columns into K-padded buffer (pad -> 0)
__global__ void gather_cols(const _Float16* __restrict__ src, const int* __restrict__ idx,
                            _Float16* __restrict__ dst, int C, int Nsrc, int K, int Kp, long total)
{
  long i = (long)blockIdx.x * blockDim.x + threadIdx.x;
  if (i >= total) return;
  int j = (int)(i % Kp);
  long t = i / Kp;
  int c = (int)(t % C);
  int b = (int)(t / C);
  _Float16 v = (_Float16)0.f;
  if (j < K) v = src[((long)b * C + c) * Nsrc + idx[(long)b * K + j]];
  dst[i] = v;
}

// gather 3D points: out[b][j][d] = pts[b][d][idx[b][j]]
__global__ void gather_pts(const float* __restrict__ pts, const int* __restrict__ idx,
                           float* __restrict__ out, int N, int K, long total)
{
  long i = (long)blockIdx.x * blockDim.x + threadIdx.x;
  if (i >= total) return;
  int d = (int)(i % 3);
  long t = i / 3;
  int j = (int)(t % K);
  int b = (int)(t / K);
  out[i] = pts[((long)b * 3 + d) * N + idx[(long)b * K + j]];
}

// softmax over keypoint similarity row + weighted sum with tgt_key -> src_corr
__global__ void softmax_corr(const float* __restrict__ S3, const float* __restrict__ tkey,
                             float* __restrict__ corr, int K, int Kp)
{
  __shared__ float red[TPB];
  __shared__ float res[4];
  const int j = blockIdx.x, b = blockIdx.y, tid = threadIdx.x;
  const float* row = S3 + ((long)b * Kp + j) * Kp;
  float mx = -3.4e38f;
  for (int t = tid; t < K; t += TPB) mx = fmaxf(mx, row[t]);
  red[tid] = mx; __syncthreads();
  for (int s = TPB / 2; s > 0; s >>= 1) { if (tid < s) red[tid] = fmaxf(red[tid], red[tid + s]); __syncthreads(); }
  mx = red[0]; __syncthreads();
  float acc[4] = {0.f, 0.f, 0.f, 0.f};
  for (int t = tid; t < K; t += TPB) {
    float e = __expf(row[t] - mx);
    const float* tk = tkey + ((long)b * K + t) * 3;
    acc[0] += e; acc[1] += e * tk[0]; acc[2] += e * tk[1]; acc[3] += e * tk[2];
  }
  for (int q = 0; q < 4; ++q) {
    red[tid] = acc[q]; __syncthreads();
    for (int s = TPB / 2; s > 0; s >>= 1) { if (tid < s) red[tid] += red[tid + s]; __syncthreads(); }
    if (tid == 0) res[q] = red[0];
    __syncthreads();
  }
  if (tid < 3) corr[((long)b * K + j) * 3 + tid] = res[tid + 1] / res[0];
}

__device__ inline float det3f(const float m[3][3]) {
  return m[0][0] * (m[1][1] * m[2][2] - m[1][2] * m[2][1])
       - m[0][1] * (m[1][0] * m[2][2] - m[1][2] * m[2][0])
       + m[0][2] * (m[1][0] * m[2][1] - m[1][1] * m[2][0]);
}
__device__ inline void cross3(const float a[3], const float b[3], float o[3]) {
  o[0] = a[1] * b[2] - a[2] * b[1];
  o[1] = a[2] * b[0] - a[0] * b[2];
  o[2] = a[0] * b[1] - a[1] * b[0];
}

// per-batch Kabsch: means, H, 3x3 SVD (Jacobi on H^T H), R & t
__global__ void kabsch(const float* __restrict__ skey, const float* __restrict__ corr,
                       float* __restrict__ out, int K, int Bn)
{
  __shared__ float red[TPB];
  __shared__ float tot[15];
  const int b = blockIdx.x, tid = threadIdx.x;
  float acc[15];
  for (int q = 0; q < 15; ++q) acc[q] = 0.f;
  for (int i = tid; i < K; i += TPB) {
    const float* p = skey + ((long)b * K + i) * 3;
    const float* q = corr + ((long)b * K + i) * 3;
    acc[0] += p[0]; acc[1] += p[1]; acc[2] += p[2];
    acc[3] += q[0]; acc[4] += q[1]; acc[5] += q[2];
    acc[6]  += p[0] * q[0]; acc[7]  += p[0] * q[1]; acc[8]  += p[0] * q[2];
    acc[9]  += p[1] * q[0]; acc[10] += p[1] * q[1]; acc[11] += p[1] * q[2];
    acc[12] += p[2] * q[0]; acc[13] += p[2] * q[1]; acc[14] += p[2] * q[2];
  }
  for (int q = 0; q < 15; ++q) {
    red[tid] = acc[q]; __syncthreads();
    for (int s = TPB / 2; s > 0; s >>= 1) { if (tid < s) red[tid] += red[tid + s]; __syncthreads(); }
    if (tid == 0) tot[q] = red[0];
    __syncthreads();
  }
  if (tid != 0) return;
  float invK = 1.f / (float)K;
  float sm[3], cm[3], H[3][3];
  for (int d = 0; d < 3; ++d) { sm[d] = tot[d] * invK; cm[d] = tot[3 + d] * invK; }
  for (int i = 0; i < 3; ++i)
    for (int j = 0; j < 3; ++j)
      H[i][j] = tot[6 + i * 3 + j] - (float)K * sm[i] * cm[j];
  float Sm[3][3], V[3][3] = {{1,0,0},{0,1,0},{0,0,1}};
  for (int i = 0; i < 3; ++i)
    for (int j = 0; j < 3; ++j)
      Sm[i][j] = H[0][i] * H[0][j] + H[1][i] * H[1][j] + H[2][i] * H[2][j];
  const int PP[3] = {0, 0, 1}, QQ[3] = {1, 2, 2};
  for (int sweep = 0; sweep < 12; ++sweep)
    for (int pi = 0; pi < 3; ++pi) {
      int p = PP[pi], q = QQ[pi];
      float apq = Sm[p][q];
      if (fabsf(apq) < 1e-30f) continue;
      float th = (Sm[q][q] - Sm[p][p]) / (2.f * apq);
      float t = copysignf(1.f, th) / (fabsf(th) + sqrtf(th * th + 1.f));
      float c = rsqrtf(t * t + 1.f), s = t * c;
      for (int k = 0; k < 3; ++k) { float a = Sm[k][p], bb = Sm[k][q]; Sm[k][p] = c * a - s * bb; Sm[k][q] = s * a + c * bb; }
      for (int k = 0; k < 3; ++k) { float a = Sm[p][k], bb = Sm[q][k]; Sm[p][k] = c * a - s * bb; Sm[q][k] = s * a + c * bb; }
      for (int k = 0; k < 3; ++k) { float a = V[k][p],  bb = V[k][q];  V[k][p]  = c * a - s * bb; V[k][q]  = s * a + c * bb; }
    }
  float ev[3] = {Sm[0][0], Sm[1][1], Sm[2][2]};
  for (int a = 0; a < 2; ++a)
    for (int c2 = a + 1; c2 < 3; ++c2)
      if (ev[c2] > ev[a]) {
        float tv = ev[a]; ev[a] = ev[c2]; ev[c2] = tv;
        for (int k = 0; k < 3; ++k) { float w = V[k][a]; V[k][a] = V[k][c2]; V[k][c2] = w; }
      }
  float u0[3], u1[3], u2[3];
  for (int k = 0; k < 3; ++k) u0[k] = H[k][0] * V[0][0] + H[k][1] * V[1][0] + H[k][2] * V[2][0];
  float n0 = sqrtf(u0[0]*u0[0] + u0[1]*u0[1] + u0[2]*u0[2]);
  if (n0 > 1e-12f) { for (int k = 0; k < 3; ++k) u0[k] /= n0; }
  else { u0[0] = 1.f; u0[1] = 0.f; u0[2] = 0.f; }
  for (int k = 0; k < 3; ++k) u1[k] = H[k][0] * V[0][1] + H[k][1] * V[1][1] + H[k][2] * V[2][1];
  float dp = u1[0]*u0[0] + u1[1]*u0[1] + u1[2]*u0[2];
  for (int k = 0; k < 3; ++k) u1[k] -= dp * u0[k];
  float n1 = sqrtf(u1[0]*u1[0] + u1[1]*u1[1] + u1[2]*u1[2]);
  if (n1 > 1e-12f) { for (int k = 0; k < 3; ++k) u1[k] /= n1; }
  else {
    float ax[3] = {1.f, 0.f, 0.f};
    if (fabsf(u0[0]) > 0.9f) { ax[0] = 0.f; ax[1] = 1.f; }
    cross3(u0, ax, u1);
    float nn = rsqrtf(u1[0]*u1[0] + u1[1]*u1[1] + u1[2]*u1[2]);
    for (int k = 0; k < 3; ++k) u1[k] *= nn;
  }
  cross3(u0, u1, u2);
  float U[3][3];
  for (int k = 0; k < 3; ++k) { U[k][0] = u0[k]; U[k][1] = u1[k]; U[k][2] = u2[k]; }
  float r[3][3];
  for (int i = 0; i < 3; ++i)
    for (int k = 0; k < 3; ++k)
      r[i][k] = V[i][0] * U[k][0] + V[i][1] * U[k][1] + V[i][2] * U[k][2];
  float d = det3f(r);
  float cd[3] = {1.f, 1.f, d};
  float R[3][3];
  for (int i = 0; i < 3; ++i)
    for (int k = 0; k < 3; ++k)
      R[i][k] = V[i][0] * cd[0] * U[k][0] + V[i][1] * cd[1] * U[k][1] + V[i][2] * cd[2] * U[k][2];
  for (int i = 0; i < 3; ++i)
    for (int k = 0; k < 3; ++k)
      out[(long)b * 9 + i * 3 + k] = R[i][k];
  for (int i = 0; i < 3; ++i)
    out[(long)Bn * 9 + (long)b * 3 + i] = cm[i] - (R[i][0] * sm[0] + R[i][1] * sm[1] + R[i][2] * sm[2]);
}

// ---------------------------------------------------------------------------
extern "C" void kernel_launch(void* const* d_in, const int* in_sizes, int n_in,
                              void* d_out, int out_size, void* d_ws, size_t ws_size,
                              hipStream_t stream)
{
  (void)n_in; (void)out_size; (void)ws_size;
  const int B = 4, C1 = 512, Cc = 1024, Fh = 2048;
  const int N  = in_sizes[0] / (B * 3);    // 2048
  const int K  = (int)(N * 0.9);           // 1843
  const int Kp = (K + 127) & ~127;         // 1920: % 128 (M-tile) and % 64 (N-tile)

  const float* src     = (const float*)d_in[0];
  const float* tgt     = (const float*)d_in[1];
  const float* src_emb = (const float*)d_in[2];
  const float* tgt_emb = (const float*)d_in[3];
  const float* sw1 = (const float*)d_in[4];
  const float* sg1 = (const float*)d_in[6];  const float* sbe1 = (const float*)d_in[7];
  const float* sw2 = (const float*)d_in[8];
  const float* sg2 = (const float*)d_in[10]; const float* sbe2 = (const float*)d_in[11];
  const float* tw1 = (const float*)d_in[12];
  const float* tg1 = (const float*)d_in[14]; const float* tbe1 = (const float*)d_in[15];
  const float* tw2 = (const float*)d_in[16];
  const float* tg2 = (const float*)d_in[18]; const float* tbe2 = (const float*)d_in[19];
  // sb1/sb2/tb1/tb2 (d_in[5,9,13,17]) cancel exactly under batch-stat BN.

  const long C1N = (long)C1 * N, CN = (long)Cc * N, NN = (long)N * N;
  const long FhN = (long)Fh * N, CKp = (long)Cc * Kp;

  char* ws = (char*)d_ws;
  size_t off = 0;
  auto take = [&](size_t bytes) -> char* {
    char* p = ws + off; off += (bytes + 255) & ~(size_t)255; return p;
  };
  _Float16* se_raw16 = (_Float16*)take((size_t)B * C1N * 2);
  _Float16* te_raw16 = (_Float16*)take((size_t)B * C1N * 2);
  _Float16* se_n16   = (_Float16*)take((size_t)B * C1N * 2);
  _Float16* te_n16   = (_Float16*)take((size_t)B * C1N * 2);
  float*    Sbuf     = (float*)   take((size_t)B * NN * 4);   // S1 / S2 / S3
  char*     P1       =            take((size_t)B * CN * 4);   // simi1_f16 -> y1 -> se_f
  char*     P2       =            take((size_t)B * CN * 4);   // simi2_f16 -> y2
  char*     G1       =            take((size_t)B * C1N * 4);  // gt1 -> h16
  char*     G2       =            take((size_t)B * C1N * 4);  // gt2
  _Float16* F1       = (_Float16*)take((size_t)B * FhN * 2);
  _Float16* F2       = (_Float16*)take((size_t)B * FhN * 2);
  _Float16* FN1      = (_Float16*)take((size_t)B * CN * 2);   // normalized se_f (f16)
  _Float16* FN2      = (_Float16*)take((size_t)B * CN * 2);
  _Float16* KEY1     = (_Float16*)take((size_t)B * CKp * 2);
  _Float16* KEY2     = (_Float16*)take((size_t)B * CKp * 2);
  _Float16* sw1_16   = (_Float16*)take((size_t)Cc * Fh * 2);
  _Float16* sw2_16   = (_Float16*)take((size_t)Cc * Cc * 2);
  _Float16* tw1_16   = (_Float16*)take((size_t)Cc * Fh * 2);
  _Float16* tw2_16   = (_Float16*)take((size_t)Cc * Cc * 2);
  float*    gmax1    = (float*)take((size_t)B * C1 * 4);
  float*    gmax2    = (float*)take((size_t)B * C1 * 4);
  float*    muA      = (float*)take((size_t)Cc * 4);
  float*    varA     = (float*)take((size_t)Cc * 4);
  float*    score1   = (float*)take((size_t)B * N * 4);
  float*    score2   = (float*)take((size_t)B * N * 4);
  int*      idx1     = (int*)  take((size_t)B * K * 4);
  int*      idx2     = (int*)  take((size_t)B * K * 4);
  float*    skey     = (float*)take((size_t)B * K * 3 * 4);
  float*    tkey     = (float*)take((size_t)B * K * 3 * 4);
  float*    corr     = (float*)take((size_t)B * K * 3 * 4);

  auto gemm = [&](const _Float16* A, const _Float16* Bm, float* Cm, int M, int Nn, int Kk,
                  long sam, long sak, long ab, long sbk, long sbn, long bb,
                  long scm, long scn, long cb) {
    dim3 g(M / 128, Nn / 64, B);
    dim3 blk(32, 8);
    wmma_gemm<<<g, blk, 0, stream>>>(A, Bm, Cm, M, Nn, Kk,
                                     sam, sak, ab, sbk, sbn, bb, scm, scn, cb);
  };
  auto cdiv = [](long a, long b) { return (int)((a + b - 1) / b); };

  // 0) weights -> f16
  f32_to_f16<<<cdiv((long)Cc * Fh, 256), 256, 0, stream>>>(sw1, sw1_16, (long)Cc * Fh);
  f32_to_f16<<<cdiv((long)Cc * Cc, 256), 256, 0, stream>>>(sw2, sw2_16, (long)Cc * Cc);
  f32_to_f16<<<cdiv((long)Cc * Fh, 256), 256, 0, stream>>>(tw1, tw1_16, (long)Cc * Fh);
  f32_to_f16<<<cdiv((long)Cc * Cc, 256), 256, 0, stream>>>(tw2, tw2_16, (long)Cc * Cc);

  // 1) normalize embeddings (raw f16 + unit-norm f16)
  int totBN = B * N;
  colnorm<<<cdiv(totBN, 256), 256, 0, stream>>>(src_emb, se_raw16, se_n16, C1, N, totBN);
  colnorm<<<cdiv(totBN, 256), 256, 0, stream>>>(tgt_emb, te_raw16, te_n16, C1, N, totBN);

  // 2) S1 = cos(se,te): (N x N), K=C1.  cos(te,se) == S1^T (computed once)
  gemm(se_n16, te_n16, Sbuf, N, N, C1, 1, N, C1N, N, 1, C1N, N, 1, NN);

  // 3) simi1 = softmax_rows(S1), simi2 = softmax_cols(S1) (both f16)
  dim3 gRows(N, B);
  softmax_f16<<<gRows, TPB, 0, stream>>>(Sbuf, (_Float16*)P1, N, N, 1, NN, NN);
  softmax_f16<<<gRows, TPB, 0, stream>>>(Sbuf, (_Float16*)P2, N, 1, N, NN, NN);

  // 4) per-channel max of raw embeddings (glob_src)
  dim3 gC1(C1, B);
  rowmax<<<gC1, TPB, 0, stream>>>(src_emb, gmax1, N, N, 1, C1N, C1);
  rowmax<<<gC1, TPB, 0, stream>>>(tgt_emb, gmax2, N, N, 1, C1N, C1);

  // 5) glob_tar = simi @ emb^T  -> stored (C1, N) per batch
  gemm((_Float16*)P1, te_raw16, (float*)G1, N, C1, N, N, 1, NN, 1, N, C1N, 1, N, C1N);
  gemm((_Float16*)P2, se_raw16, (float*)G2, N, C1, N, N, 1, NN, 1, N, C1N, 1, N, C1N);

  // 6) assemble feature-interaction tensors (f16)
  long totF = (long)B * Fh * N;
  assemble_feat<<<cdiv(totF, 256), 256, 0, stream>>>(se_raw16, (float*)G1, gmax1, F1, C1, N, totF);
  assemble_feat<<<cdiv(totF, 256), 256, 0, stream>>>(te_raw16, (float*)G2, gmax2, F2, C1, N, totF);

  // 7) emb_nn2 per branch (conv1+BN+LReLU, conv2+BN+LReLU, L2-normalize)
  auto branch = [&](const _Float16* feat, const _Float16* w1f, const float* g1, const float* be1,
                    const _Float16* w2f, const float* g2, const float* be2, _Float16* outn16) {
    float* y1 = (float*)P1;
    float* y2 = (float*)P2;
    _Float16* h16 = (_Float16*)G1;
    float* f32o = (float*)P1;
    long totC = (long)B * Cc * N;
    gemm(w1f, feat, y1, Cc, N, Fh, Fh, 1, 0, N, 1, FhN, N, 1, CN);
    bn_stats<<<Cc, TPB, 0, stream>>>(y1, muA, varA, B, Cc, N);
    bn_lrelu<<<cdiv(totC, 256), 256, 0, stream>>>(y1, muA, varA, g1, be1, h16, nullptr, Cc, N, totC);
    gemm(w2f, h16, y2, Cc, N, Cc, Cc, 1, 0, N, 1, CN, N, 1, CN);
    bn_stats<<<Cc, TPB, 0, stream>>>(y2, muA, varA, B, Cc, N);
    bn_lrelu<<<cdiv(totC, 256), 256, 0, stream>>>(y2, muA, varA, g2, be2, nullptr, f32o, Cc, N, totC);
    colnorm<<<cdiv(totBN, 256), 256, 0, stream>>>(f32o, nullptr, outn16, Cc, N, totBN);
  };
  branch(F1, sw1_16, sg1, sbe1, sw2_16, sg2, sbe2, FN1);
  branch(F2, tw1_16, tg1, tbe1, tw2_16, tg2, tbe2, FN2);

  // 8) S2 = cos(se_f, te_f): (N x N), K=Cc
  gemm(FN1, FN2, Sbuf, N, N, Cc, 1, N, CN, N, 1, CN, N, 1, NN);

  // 9) scores = row / col max of S2
  rowmax<<<gRows, TPB, 0, stream>>>(Sbuf, score1, N, N, 1, NN, N);
  rowmax<<<gRows, TPB, 0, stream>>>(Sbuf, score2, N, 1, N, NN, N);

  // 10) top-K keypoint selection
  topk_sort<<<B, 1024, 0, stream>>>(score1, idx1, N, K);
  topk_sort<<<B, 1024, 0, stream>>>(score2, idx2, N, K);

  // 11) gathers (embedding cols zero-padded to Kp; points)
  long totG = (long)B * Cc * Kp;
  gather_cols<<<cdiv(totG, 256), 256, 0, stream>>>(FN1, idx1, KEY1, Cc, N, K, Kp, totG);
  gather_cols<<<cdiv(totG, 256), 256, 0, stream>>>(FN2, idx2, KEY2, Cc, N, K, Kp, totG);
  long totP = (long)B * K * 3;
  gather_pts<<<cdiv(totP, 256), 256, 0, stream>>>(src, idx1, skey, N, K, totP);
  gather_pts<<<cdiv(totP, 256), 256, 0, stream>>>(tgt, idx2, tkey, N, K, totP);

  // 12) S3 = cos(se_key, te_key): (Kp x Kp), K=Cc
  gemm(KEY1, KEY2, Sbuf, Kp, Kp, Cc, 1, Kp, CKp, Kp, 1, CKp, Kp, 1, (long)Kp * Kp);

  // 13) softmax + correspondence
  dim3 gK(K, B);
  softmax_corr<<<gK, TPB, 0, stream>>>(Sbuf, tkey, corr, K, Kp);

  // 14) Kabsch / SVD -> R (B,3,3) then t (B,3) into d_out
  kabsch<<<B, TPB, 0, stream>>>(skey, corr, (float*)d_out, K, B);
}
